// LoRATACMLP4_84988812853867
// MI455X (gfx1250) — compile-verified, matrix operand order checked by reading, not run on hardware
//
#include <hip/hip_runtime.h>
#include <math.h>

// ---------------- problem constants ----------------
#define B_  16
#define K_  8
#define F_  512
#define C_  512          // INCH
#define HC_ 256          // INCH/2
#define O_  512          // OUTCH
#define R_  8

typedef __bf16 bf16;
typedef __attribute__((ext_vector_type(16))) bf16 v16bf;
typedef __attribute__((ext_vector_type(8)))  float v8f;

union Frag {
    v16bf v;
    uint4 q[2];
};

// Load a 16x32 bf16 WMMA fragment (A layout; B uses same per-lane mapping with
// row = N because W is stored [out,in] so B[k,n] = W[n,k]).
// Per ISA 7.12.2: lane L (m = L&15, g = L>>4) holds k0+8g..+7 in VGPR0-3 and
// k0+16+8g..+7 in VGPR4-7 -> two contiguous 16B chunks per lane.
__device__ __forceinline__ v16bf load_frag(const bf16* __restrict__ base, int ld,
                                           int row0, int k0, int lane) {
    const int m = lane & 15;
    const int g = lane >> 4;
    const bf16* p = base + (long)(row0 + m) * ld + k0 + (g << 3);
    Frag f;
    f.q[0] = *(const uint4*)(p);
    f.q[1] = *(const uint4*)(p + 16);
    return f.v;
}

__device__ __forceinline__ float gelu_exact(float x) {
    return 0.5f * x * (1.0f + erff(x * 0.70710678118654752440f));
}

// ---------------- kernel 1: weights fp32 -> bf16 ----------------
__global__ __launch_bounds__(256) void convert_w(const float* __restrict__ Wp,
                                                 const float* __restrict__ Wa,
                                                 const float* __restrict__ Wo,
                                                 bf16* __restrict__ bWp,
                                                 bf16* __restrict__ bWa,
                                                 bf16* __restrict__ bWo) {
    int id = blockIdx.x * 256 + threadIdx.x;   // 524288 total
    if (id < HC_ * C_) {
        bWp[id] = (bf16)Wp[id];
    } else if (id < 2 * HC_ * C_) {
        int j = id - HC_ * C_;
        bWa[j] = (bf16)Wa[j];
    } else {
        int j = id - 2 * HC_ * C_;
        if (j < O_ * C_) bWo[j] = (bf16)Wo[j];
    }
}

// ---------------- kernel 2: xbar = mean_k x, bf16 ----------------
__global__ __launch_bounds__(256) void reduce_x(const float* __restrict__ x,
                                                bf16* __restrict__ xbar) {
    long id = (long)blockIdx.x * 256 + threadIdx.x;  // B*F*C = 4194304
    if (id >= (long)B_ * F_ * C_) return;
    int i = (int)(id % C_);
    int f = (int)((id / C_) % F_);
    int b = (int)(id / ((long)F_ * C_));
    const float* p = x + (((long)b * K_ * F_) + f) * C_ + i;
    float s = 0.f;
    #pragma unroll
    for (int k = 0; k < K_; ++k) s += p[(long)k * F_ * C_];
    xbar[id] = (bf16)(s * (1.0f / K_));
}

// ---------------- kernel 3: ybar = gelu(xbar @ W_ave^T + b_ave), bf16 -------
// M = B*F = 8192, N = 256, Kdim = 512. One 16x16 tile per wave (small GEMM).
__global__ __launch_bounds__(256) void ave_gemm(const bf16* __restrict__ xbar,
                                                const bf16* __restrict__ Wa,
                                                const float* __restrict__ b_ave,
                                                bf16* __restrict__ ybar) {
    const int lane = threadIdx.x & 31;
    const int wid  = blockIdx.x * 8 + (threadIdx.x >> 5);   // 8192 waves
    const int m0 = (wid >> 4) * 16;        // 512 M-tiles
    const int n0 = (wid & 15) * 16;        // 16  N-tiles
    v8f acc = {};
    #pragma unroll
    for (int kt = 0; kt < 16; ++kt) {
        v16bf a  = load_frag(xbar, C_, m0, kt * 32, lane);
        v16bf bm = load_frag(Wa,   C_, n0, kt * 32, lane);
        acc = __builtin_amdgcn_wmma_f32_16x16x32_bf16(false, a, false, bm,
                                                      (short)0, acc, false, false);
    }
    const int n  = n0 + (lane & 15);
    const int mb = m0 + 8 * (lane >> 4);
    const float ba = b_ave[n];
    #pragma unroll
    for (int i = 0; i < 8; ++i)
        ybar[(long)(mb + i) * HC_ + n] = (bf16)gelu_exact(acc[i] + ba);
}

// ---------------- kernel 4: fused pass-GEMM + gelu + LoRA + out-GEMM --------
// One workgroup (8 waves) per (b, k, 64-row f-block). 1024 workgroups.
// Each wave owns a 16-row M-strip and accumulates 4 N-tiles per A-fragment
// (1 LDS A-load amortized over 4 WMMAs).
__global__ __launch_bounds__(256) void fused(const float* __restrict__ x,
                                             const float* __restrict__ u,
                                             const float* __restrict__ v,
                                             const float* __restrict__ bB,
                                             const float* __restrict__ b_pass,
                                             const bf16*  __restrict__ Wp,
                                             const float* __restrict__ b_out,
                                             const bf16*  __restrict__ Wo,
                                             const bf16*  __restrict__ ybar,
                                             float* __restrict__ out) {
    __shared__ bf16  Xs[64][C_];     // 64 KB: x block in bf16
    __shared__ bf16  Hs[64][C_];     // 64 KB: h = [gelu(pass), ybar]
    __shared__ float Zs[64][R_];     // 2 KB : z = h.v / inch

    const int tid  = threadIdx.x;
    const int lane = tid & 31;
    const int wave = tid >> 5;
    const int blk  = blockIdx.x;             // 0..1023
    const int fblk = blk & 7;
    const int kk   = (blk >> 3) & 7;
    const int bb   = blk >> 6;
    const int fbase = fblk * 64;
    const long rowbase = ((long)(bb * K_ + kk) * F_ + fbase);

    // ---- stage X (fp32 -> bf16) into LDS ----
    const float* xsrc = x + rowbase * C_;
    for (int i = tid; i < 64 * C_ / 4; i += 256) {
        float4 f4 = ((const float4*)xsrc)[i];
        bf16* d = &Xs[0][0] + i * 4;
        d[0] = (bf16)f4.x; d[1] = (bf16)f4.y; d[2] = (bf16)f4.z; d[3] = (bf16)f4.w;
    }
    // ---- stage ybar into Hs[:, 256:512] (depends on (b,f) only) ----
    const bf16* ysrc = ybar + ((long)bb * F_ + fbase) * HC_;
    for (int i = tid; i < 64 * HC_ / 8; i += 256) {
        int r = i >> 5;                 // 32 uint4 per 256-wide row
        int c = (i & 31) * 8;
        *(uint4*)&Hs[r][HC_ + c] = ((const uint4*)ysrc)[i];
    }
    __syncthreads();

    const int m0 = (wave & 3) * 16;

    // ---- pass GEMM: Hs[:,0:256] = gelu(Xs @ Wp^T + b_pass) ----
    // wave covers m-strip x 128 cols as 2 chunks of 4 N-tiles.
    {
        const int nhalf = (wave >> 2) * 128;
        for (int c = 0; c < 2; ++c) {
            const int nc = nhalf + c * 64;
            v8f acc0 = {}, acc1 = {}, acc2 = {}, acc3 = {};
            #pragma unroll
            for (int kt = 0; kt < 16; ++kt) {
                const int k0 = kt * 32;
                v16bf a  = load_frag(&Xs[0][0], C_, m0, k0, lane);
                v16bf b0 = load_frag(Wp, C_, nc +  0, k0, lane);
                v16bf b1 = load_frag(Wp, C_, nc + 16, k0, lane);
                v16bf b2 = load_frag(Wp, C_, nc + 32, k0, lane);
                v16bf b3 = load_frag(Wp, C_, nc + 48, k0, lane);
                acc0 = __builtin_amdgcn_wmma_f32_16x16x32_bf16(false, a, false, b0, (short)0, acc0, false, false);
                acc1 = __builtin_amdgcn_wmma_f32_16x16x32_bf16(false, a, false, b1, (short)0, acc1, false, false);
                acc2 = __builtin_amdgcn_wmma_f32_16x16x32_bf16(false, a, false, b2, (short)0, acc2, false, false);
                acc3 = __builtin_amdgcn_wmma_f32_16x16x32_bf16(false, a, false, b3, (short)0, acc3, false, false);
            }
            const int nl = lane & 15;
            const int mb = m0 + 8 * (lane >> 4);
            v8f accs[4] = {acc0, acc1, acc2, acc3};
            #pragma unroll
            for (int j = 0; j < 4; ++j) {
                const int n = nc + j * 16 + nl;
                const float bp = b_pass[n];
                #pragma unroll
                for (int i = 0; i < 8; ++i)
                    Hs[mb + i][n] = (bf16)gelu_exact(accs[j][i] + bp);
            }
        }
    }
    __syncthreads();

    // ---- z = (H @ v) / inch  (rank-8, VALU) ----
    {
        const int r   = tid & 7;
        const int row = tid >> 3;                       // 0..31
        const float* vp = v + (long)(bb * K_ + kk) * C_ * R_;   // [512][8]
        for (int rr = row; rr < 64; rr += 32) {
            float s = 0.f;
            #pragma unroll 8
            for (int i = 0; i < C_; ++i)
                s += (float)Hs[rr][i] * vp[i * R_ + r];
            Zs[rr][r] = s * (1.0f / C_);
        }
    }
    __syncthreads();

    // ---- out = gelu(H @ Wo^T + b_out + b + (z @ u^T)/R) ----
    // wave covers m-strip x 256 cols as 4 chunks of 4 N-tiles.
    const float* up   = u  + (long)(bb * K_ + kk) * O_ * R_;    // [512][8]
    const float* bvec = bB + (long)bb * O_;
    float* optr = out + rowbase * O_;
    const int nbase = (wave >> 2) * 256;
    for (int c = 0; c < 4; ++c) {
        const int nc = nbase + c * 64;
        // warm L2/WGP$ for the next chunk's B rows while this chunk computes
        if (c < 3) {
            const bf16* pf = Wo + (long)(nc + 64 + (lane & 15)) * C_ + (lane >> 4) * 256;
            __builtin_prefetch(pf, 0, 2);
        }
        v8f acc0 = {}, acc1 = {}, acc2 = {}, acc3 = {};
        #pragma unroll
        for (int kt = 0; kt < 16; ++kt) {
            const int k0 = kt * 32;
            v16bf a  = load_frag(&Hs[0][0], C_, m0, k0, lane);
            v16bf b0 = load_frag(Wo, C_, nc +  0, k0, lane);
            v16bf b1 = load_frag(Wo, C_, nc + 16, k0, lane);
            v16bf b2 = load_frag(Wo, C_, nc + 32, k0, lane);
            v16bf b3 = load_frag(Wo, C_, nc + 48, k0, lane);
            acc0 = __builtin_amdgcn_wmma_f32_16x16x32_bf16(false, a, false, b0, (short)0, acc0, false, false);
            acc1 = __builtin_amdgcn_wmma_f32_16x16x32_bf16(false, a, false, b1, (short)0, acc1, false, false);
            acc2 = __builtin_amdgcn_wmma_f32_16x16x32_bf16(false, a, false, b2, (short)0, acc2, false, false);
            acc3 = __builtin_amdgcn_wmma_f32_16x16x32_bf16(false, a, false, b3, (short)0, acc3, false, false);
        }
        const int nl = lane & 15;
        const int mb = m0 + 8 * (lane >> 4);
        v8f accs[4] = {acc0, acc1, acc2, acc3};
        #pragma unroll
        for (int j = 0; j < 4; ++j) {
            const int n = nc + j * 16 + nl;
            const float4 ua = *(const float4*)(up + n * R_);
            const float4 ub = *(const float4*)(up + n * R_ + 4);
            const float bo = b_out[n] + bvec[n];
            #pragma unroll
            for (int i = 0; i < 8; ++i) {
                const int m = mb + i;
                float lora = (Zs[m][0] * ua.x + Zs[m][1] * ua.y +
                              Zs[m][2] * ua.z + Zs[m][3] * ua.w +
                              Zs[m][4] * ub.x + Zs[m][5] * ub.y +
                              Zs[m][6] * ub.z + Zs[m][7] * ub.w) * (1.0f / R_);
                optr[(long)m * O_ + n] = gelu_exact(accs[j][i] + bo + lora);
            }
        }
    }
}

// ---------------- launcher ----------------
extern "C" void kernel_launch(void* const* d_in, const int* in_sizes, int n_in,
                              void* d_out, int out_size, void* d_ws, size_t ws_size,
                              hipStream_t stream) {
    const float* x  = (const float*)d_in[0];
    const float* u  = (const float*)d_in[1];
    const float* v  = (const float*)d_in[2];
    const float* bB = (const float*)d_in[3];
    const float* Wp = (const float*)d_in[4];
    const float* bp = (const float*)d_in[5];
    const float* Wa = (const float*)d_in[6];
    const float* ba = (const float*)d_in[7];
    const float* Wo = (const float*)d_in[8];
    const float* bo = (const float*)d_in[9];
    float* out = (float*)d_out;

    char* ws = (char*)d_ws;
    bf16* xbar = (bf16*)(ws);                       //  8,388,608 B
    bf16* ybar = (bf16*)(ws + 8388608);             //  4,194,304 B
    bf16* bWp  = (bf16*)(ws + 12582912);            //    262,144 B
    bf16* bWa  = (bf16*)(ws + 12845056);            //    262,144 B
    bf16* bWo  = (bf16*)(ws + 13107200);            //    524,288 B

    convert_w<<<2048, 256, 0, stream>>>(Wp, Wa, Wo, bWp, bWa, bWo);
    reduce_x<<<(B_ * F_ * C_) / 256, 256, 0, stream>>>(x, xbar);
    ave_gemm<<<1024, 256, 0, stream>>>(xbar, bWa, ba, ybar);
    fused<<<1024, 256, 0, stream>>>(x, u, v, bB, bp, bWp, bo, bWo, ybar, out);
}
